// STFT_2783138808431
// MI455X (gfx1250) — compile-verified, matrix operand order checked by reading, not run on hardware
//
#include <hip/hip_runtime.h>
#include <hip/hip_bf16.h>
#include <math.h>

#define FILT     1024
#define HOP      256
#define CUT      513
#define NBATCH   32
#define LSIG     262144
#define NFRAMES  1025
#define FRPAD    1088      // 17 * 64 padded frames
#define CHP      544       // 17 * 32 padded half-channels (re or im)
#define KPAD     1088      // 2 * CHP, GEMM2 K dim
#define OLALEN   263168    // FILT + HOP*(NFRAMES-1)
#define STEP     6.1359231515425649e-03f  // 2*pi/1024

typedef __bf16 bf16;
typedef __bf16        v16bf __attribute__((ext_vector_type(16)));
typedef float         v8f   __attribute__((ext_vector_type(8)));
typedef unsigned int  v4u   __attribute__((ext_vector_type(4)));
typedef int           v8i   __attribute__((ext_vector_type(8)));
typedef int           v4i   __attribute__((ext_vector_type(4)));

// ---------------------------------------------------------------------------
// TDM: 2D tile load global -> LDS via D# descriptor (ISA 08_async_tensor §8).
// data_size = 1 (2-byte bf16 elements); dims/strides in elements.
// amdgpu-toolchain (clang-23): 6-arg builtin form.
// ---------------------------------------------------------------------------
__device__ inline unsigned lds_off(const void* p) {
  return (unsigned)(unsigned long long)p;   // low 32 bits of generic addr = LDS offset
}

__device__ inline void tdm_load_2d(unsigned lds_addr, const void* gptr,
                                   unsigned tdim0, unsigned tdim1,
                                   unsigned tile0, unsigned tile1,
                                   unsigned stride0) {
  unsigned long long ga = (unsigned long long)gptr;
  v4u g0;
  g0[0] = 1u;                                   // count=1, is_restore=0, gather off
  g0[1] = lds_addr;                             // lds_addr (bytes)
  g0[2] = (unsigned)(ga & 0xFFFFFFFFu);         // global_addr[31:0]
  g0[3] = (unsigned)((ga >> 32) & 0x1FFFFFFu) | (2u << 30);  // addr[56:32] | type=2
  v8i g1;
  g1[0] = (1 << 16);                            // workgroup_mask=0, data_size=1 (2B)
  g1[1] = (int)((tdim0 & 0xFFFFu) << 16);       // abar=0 | tensor_dim0[15:0]
  g1[2] = (int)((tdim0 >> 16) | ((tdim1 & 0xFFFFu) << 16));
  g1[3] = (int)((tdim1 >> 16) | (tile0 << 16)); // tensor_dim1[31:16] | tile_dim0
  g1[4] = (int)tile1;                           // tile_dim1 | tile_dim2=0
  g1[5] = (int)stride0;                         // tensor_dim0_stride[31:0]
  g1[6] = 0;                                    // stride0[47:32] | stride1[15:0]
  g1[7] = 0;
  v4i z4 = {0, 0, 0, 0};
  v8i z8 = {0, 0, 0, 0, 0, 0, 0, 0};
  __builtin_amdgcn_tensor_load_to_lds(g0, g1, z4, z4, z8, 0);
}

// ---------------------------------------------------------------------------
// WMMA fragment loader, 16-bit A-layout (16x32 tile, row-major in LDS).
// lane 0-15: M=lane, K-halves {0..7,16..23}; lane 16-31: M=lane-16, {8..15,24..31}
// ---------------------------------------------------------------------------
__device__ inline v16bf load_frag(const bf16* p, int stride) {
  int lane = threadIdx.x & 31;
  int r  = lane & 15;
  int kh = (lane >> 4) << 3;
  v16bf f;
#pragma unroll
  for (int v = 0; v < 8; ++v) {
    int k = ((v >> 2) << 4) + kh + ((v & 3) << 1);
    f[2 * v]     = p[r * stride + k];
    f[2 * v + 1] = p[r * stride + k + 1];
  }
  return f;
}

// ---------------------------------------------------------------------------
// Kernel 1: generate bf16 bases on device.
//   fb  [KPAD][FILT] channel-major:  fb[c][n]  = cos/-sin(2*pi*c*n/N) * win[n]
//   ibT [FILT][KPAD] tap-major:      ibT[j][c] = 0.25*(g_c/N)*cos/-sin * win[j]
// (pinv of the real-DFT stack is a diagonal-scaled transpose: F*F^T diagonal)
// ---------------------------------------------------------------------------
__global__ void k_bases(bf16* __restrict__ fb, bf16* __restrict__ ibT) {
  int idx = blockIdx.x * blockDim.x + threadIdx.x;
  if (idx >= KPAD * FILT) return;
  int c = idx / FILT, n = idx % FILT;
  float win = 0.5f - 0.5f * cosf((float)n * STEP);
  float vf = 0.f, vi = 0.f;
  if (c < CUT) {
    int m = (c * n) & 1023;
    float g = (c == 0 || c == 512) ? 1.f : 2.f;
    float cs = cosf((float)m * STEP);
    vf = cs * win;
    vi = 0.25f * (g / 1024.f) * cs * win;
  } else if (c >= CHP && c < CHP + CUT) {
    int cc = c - CHP;
    int m = (cc * n) & 1023;
    float g = (cc == 0 || cc == 512) ? 1.f : 2.f;
    float sn = sinf((float)m * STEP);
    vf = -sn * win;
    vi = -0.25f * (g / 1024.f) * sn * win;
  }
  fb[c * FILT + n]  = (bf16)vf;
  ibT[n * KPAD + c] = (bf16)vi;
}

__global__ void k_zero(float* __restrict__ p, int n) {
  int i = blockIdx.x * blockDim.x + threadIdx.x;
  if (i < n) p[i] = 0.f;
}

// ---------------------------------------------------------------------------
// Kernel 3: forward STFT GEMM + mag/phase/recombine epilogue.
// Grid (17 nt, 17 mt, 32 batch), 256 threads = 8 waves (4 M-sub x 2 N-sub).
// WG tile: 64 frames x 32 re-channels (+ paired 32 im-channels at +CHP).
// Quad-buffered LDS, 2-stage TDM lookahead, one barrier per K-step.
// ---------------------------------------------------------------------------
__global__ void __launch_bounds__(256)
k_fwd(const float* __restrict__ x, const bf16* __restrict__ fb,
      bf16* __restrict__ rec) {
  __shared__ bf16 sA[4][64 * 32];
  __shared__ bf16 sBre[4][32 * 32];
  __shared__ bf16 sBim[4][32 * 32];

  const int batch = blockIdx.z, mt = blockIdx.y, nt = blockIdx.x;
  const int tid = threadIdx.x, w = tid >> 5;
  const int wr = w >> 1, wc = w & 1;
  const int chan0 = nt * 32;
  const float* xb = x + (size_t)batch * LSIG;

  v8f accRe = {}; v8f accIm = {};

  const int row = tid >> 2;           // 0..63 : frame row in tile
  const int c8  = (tid & 3) << 3;     // 8-sample chunk within k-slab
  const int frame_l = mt * 64 + row;

  auto stage = [&](int s) {
    const int slot = s & 3;
    const int k0 = s << 5;
    if (w == 0) {
      tdm_load_2d(lds_off(&sBre[slot][0]), fb + (size_t)chan0 * FILT + k0,
                  FILT, KPAD, 32, 32, FILT);
      tdm_load_2d(lds_off(&sBim[slot][0]), fb + (size_t)(CHP + chan0) * FILT + k0,
                  FILT, KPAD, 32, 32, FILT);
    }
    float vals[8];
    if (frame_l < NFRAMES) {
      int s0 = 256 * frame_l + k0 + c8 - 512;
#pragma unroll
      for (int i = 0; i < 8; ++i) {
        int ss = s0 + i;
        if (ss < 0) ss = -ss;
        if (ss >= LSIG) ss = 2 * LSIG - 2 - ss;
        vals[i] = xb[ss];
      }
    } else {
#pragma unroll
      for (int i = 0; i < 8; ++i) vals[i] = 0.f;
    }
#pragma unroll
    for (int i = 0; i < 8; ++i) sA[slot][row * 32 + c8 + i] = (bf16)vals[i];
  };

  stage(0);
  stage(1);

  for (int kt = 0; kt < 32; ++kt) {
    if (kt + 2 < 32) stage(kt + 2);
    if (w == 0) {
      if (kt + 2 < 32)      __builtin_amdgcn_s_wait_tensorcnt(4);
      else if (kt + 1 < 32) __builtin_amdgcn_s_wait_tensorcnt(2);
      else                  __builtin_amdgcn_s_wait_tensorcnt(0);
    }
    __syncthreads();
    const int slot = kt & 3;
    v16bf a  = load_frag(&sA[slot][(wr * 16) * 32], 32);
    v16bf br = load_frag(&sBre[slot][(wc * 16) * 32], 32);
    v16bf bi = load_frag(&sBim[slot][(wc * 16) * 32], 32);
    accRe = __builtin_amdgcn_wmma_f32_16x16x32_bf16(false, a, false, br,
                                                    (short)0, accRe, false, false);
    accIm = __builtin_amdgcn_wmma_f32_16x16x32_bf16(false, a, false, bi,
                                                    (short)0, accIm, false, false);
  }

  // Epilogue: faithful magnitude/phase roundtrip, store bf16 recombine.
  const int lane = tid & 31;
  const int Mb = (lane >> 4) * 8, N = lane & 15;
  const int cRe = chan0 + wc * 16 + N;
  bf16* recb = rec + (size_t)batch * FRPAD * KPAD;
#pragma unroll
  for (int r = 0; r < 8; ++r) {
    int f = mt * 64 + wr * 16 + Mb + r;
    float re = accRe[r], im = accIm[r];
    float mag = sqrtf(re * re + im * im);
    float ph  = atan2f(im, re);
    recb[(size_t)f * KPAD + cRe]       = (bf16)(mag * cosf(ph));
    recb[(size_t)f * KPAD + CHP + cRe] = (bf16)(mag * sinf(ph));
  }
}

// ---------------------------------------------------------------------------
// Kernel 4: inverse GEMM + overlap-add.
// Grid (32 nt over 1024 taps, 17 mt, 32 batch); WG tile 64 frames x 32 taps.
// Both A (recombine) and B (ibT) tiles via TDM, quad-buffered pipeline.
// ---------------------------------------------------------------------------
__global__ void __launch_bounds__(256)
k_inv(const bf16* __restrict__ rec, const bf16* __restrict__ ibT,
      float* __restrict__ ola) {
  __shared__ bf16 sA[4][64 * 32];
  __shared__ bf16 sB[4][32 * 32];

  const int batch = blockIdx.z, mt = blockIdx.y, nt = blockIdx.x;
  const int tid = threadIdx.x, w = tid >> 5;
  const int wr = w >> 1, wc = w & 1;
  const bf16* recb = rec + (size_t)batch * FRPAD * KPAD;

  v8f acc = {};

  auto stage = [&](int s) {
    const int slot = s & 3;
    const int k0 = s << 5;
    if (w == 0) {
      tdm_load_2d(lds_off(&sA[slot][0]), recb + (size_t)(mt * 64) * KPAD + k0,
                  KPAD, FRPAD, 32, 64, KPAD);
      tdm_load_2d(lds_off(&sB[slot][0]), ibT + (size_t)(nt * 32) * KPAD + k0,
                  KPAD, FILT, 32, 32, KPAD);
    }
  };

  stage(0);
  stage(1);

  for (int kt = 0; kt < 34; ++kt) {
    if (kt + 2 < 34) stage(kt + 2);
    if (w == 0) {
      if (kt + 2 < 34)      __builtin_amdgcn_s_wait_tensorcnt(4);
      else if (kt + 1 < 34) __builtin_amdgcn_s_wait_tensorcnt(2);
      else                  __builtin_amdgcn_s_wait_tensorcnt(0);
    }
    __syncthreads();
    const int slot = kt & 3;
    v16bf a = load_frag(&sA[slot][(wr * 16) * 32], 32);
    v16bf b = load_frag(&sB[slot][(wc * 16) * 32], 32);
    acc = __builtin_amdgcn_wmma_f32_16x16x32_bf16(false, a, false, b,
                                                  (short)0, acc, false, false);
  }

  const int lane = tid & 31;
  const int Mb = (lane >> 4) * 8, N = lane & 15;
  const int j = nt * 32 + wc * 16 + N;
  float* olab = ola + (size_t)batch * OLALEN;
#pragma unroll
  for (int r = 0; r < 8; ++r) {
    int f = mt * 64 + wr * 16 + Mb + r;
    if (f < NFRAMES) atomicAdd(&olab[256 * f + j], acc[r]);
  }
}

// ---------------------------------------------------------------------------
// Kernel 5: window-sumsquare normalization (analytic), scale, crop.
// ---------------------------------------------------------------------------
__global__ void k_final(const float* __restrict__ ola, float* __restrict__ out) {
  int idx = blockIdx.x * blockDim.x + threadIdx.x;
  if (idx >= NBATCH * LSIG) return;
  int b = idx / LSIG, t = idx % LSIG;
  int tp = t + 512;
  int hi = tp >> 8; if (hi > NFRAMES - 1) hi = NFRAMES - 1;
  int lo = tp - 1023;
  int i0 = lo > 0 ? ((lo + 255) >> 8) : 0;
  float wsum = 0.f;
  for (int i = i0; i <= hi; ++i) {
    int m = tp - (i << 8);
    float h = 0.5f - 0.5f * cosf((float)m * STEP);
    wsum += h * h;
  }
  float v = ola[(size_t)b * OLALEN + tp];
  out[idx] = (wsum > 1.1754944e-38f) ? v * (4.0f / wsum) : v * 4.0f;
}

// ---------------------------------------------------------------------------
extern "C" void kernel_launch(void* const* d_in, const int* in_sizes, int n_in,
                              void* d_out, int out_size, void* d_ws, size_t ws_size,
                              hipStream_t stream) {
  const float* x = (const float*)d_in[0];
  float* out = (float*)d_out;
  char* ws = (char*)d_ws;

  const size_t sz_fb  = (size_t)KPAD * FILT * 2;            // 2,228,224
  const size_t sz_ib  = (size_t)FILT * KPAD * 2;            // 2,228,224
  const size_t sz_rec = (size_t)NBATCH * FRPAD * KPAD * 2;  // 75,759,616
  bf16*  fb  = (bf16*)(ws);
  bf16*  ibT = (bf16*)(ws + sz_fb);
  bf16*  rec = (bf16*)(ws + sz_fb + sz_ib);
  float* ola = (float*)(ws + sz_fb + sz_ib + sz_rec);

  k_bases<<<(KPAD * FILT + 255) / 256, 256, 0, stream>>>(fb, ibT);
  k_zero<<<(NBATCH * OLALEN + 255) / 256, 256, 0, stream>>>(ola, NBATCH * OLALEN);

  dim3 g1(17, 17, NBATCH);
  k_fwd<<<g1, 256, 0, stream>>>(x, fb, rec);

  dim3 g2(32, 17, NBATCH);
  k_inv<<<g2, 256, 0, stream>>>(rec, ibT, ola);

  k_final<<<(NBATCH * LSIG + 255) / 256, 256, 0, stream>>>(ola, out);
}
